// MultiHeadAttention_22840636080982
// MI455X (gfx1250) — matrix operand order repaired and checked by reading, not compiled, not run on hardware
//
#include <hip/hip_runtime.h>
#include <hip/hip_bf16.h>

typedef __bf16 bf16_t;
typedef float  v8f  __attribute__((ext_vector_type(8)));
typedef bf16_t v8bf __attribute__((ext_vector_type(8)));
typedef bf16_t v16bf __attribute__((ext_vector_type(16)));

static constexpr int D_MODEL = 1024;
static constexpr int S_LEN   = 2048;
static constexpr int NHEAD   = 16;
static constexpr int DK      = 64;
static constexpr int BATCH   = 2;
static constexpr int MROWS   = BATCH * S_LEN;   // 4096

__device__ __forceinline__ v16bf cat8(v8bf lo, v8bf hi) {
  return __builtin_shufflevector(lo, hi, 0,1,2,3,4,5,6,7,8,9,10,11,12,13,14,15);
}

// Async DMA: 16 bytes global -> LDS, tracked by ASYNCcnt (CDNA5).
__device__ __forceinline__ void async_copy_b128(void* lds_dst, const void* gsrc) {
  const unsigned lds_off = (unsigned)(uintptr_t)lds_dst;   // low 32 bits = LDS offset
  asm volatile("global_load_async_to_lds_b128 %0, %1, off"
               :: "v"(lds_off), "v"((unsigned long long)(uintptr_t)gsrc)
               : "memory");
}
__device__ __forceinline__ void wait_async0() {
  asm volatile("s_wait_asynccnt 0x0" ::: "memory");
}

// ---------------------------------------------------------------------------
// GEMM: Y[M,N] = A[M,K] @ W[K,N]; A is f32 or bf16, Y is bf16 or f32.
// Block tile 128x64, 8 waves (4x2), each wave 32x32 (2x2 WMMA), K step 32.
// LDS rows padded to 40 bf16 (80B) -> conflict-free b128 fragment reads.
// bf16 A tiles are staged with global_load_async_to_lds_b128.
// ---------------------------------------------------------------------------
template<bool A_BF16, bool OUT_BF16>
__global__ __launch_bounds__(256)
void gemm_kernel(const void* __restrict__ Av, const float* __restrict__ W,
                 void* __restrict__ Yv, int M, int K, int N)
{
  __shared__ __align__(16) bf16_t As[128][40];
  __shared__ __align__(16) bf16_t Bt[64][40];   // transposed W tile: Bt[n][k]

  const int tid  = threadIdx.x;
  const int lane = tid & 31, wave = tid >> 5;
  const int wm = wave >> 1, wn = wave & 1;      // 4x2 wave grid
  const int lq = lane & 15, half = lane >> 4;
  const int m0 = blockIdx.x * 128, n0 = blockIdx.y * 64;

  const float*  Af = (const float*)Av;
  const bf16_t* Ab = (const bf16_t*)Av;

  const int arow = tid >> 1;              // 0..127
  const int acol = (tid & 1) * 16;        // 0 or 16
  const int brow = tid >> 3;              // k: 0..31
  const int bcol = (tid & 7) * 8;         // n: 0..56

  v8f c[2][2] = {};

  for (int k0 = 0; k0 < K; k0 += 32) {
    // ---- stage A tile ----
    if (A_BF16) {
      const bf16_t* src = Ab + (size_t)(m0 + arow) * K + k0 + acol;
      async_copy_b128(&As[arow][acol], src);        // DMA, no VGPR round trip
    } else {
      const float* src = Af + (size_t)(m0 + arow) * K + k0 + acol;
      v8bf x0, x1;
      #pragma unroll
      for (int i = 0; i < 8; ++i) { x0[i] = (bf16_t)src[i]; x1[i] = (bf16_t)src[8 + i]; }
      *(v8bf*)&As[arow][acol]     = x0;
      *(v8bf*)&As[arow][acol + 8] = x1;
    }
    // ---- stage W tile transposed (f32 -> bf16 convert) ----
    {
      const float* wsrc = W + (size_t)(k0 + brow) * N + n0 + bcol;
      #pragma unroll
      for (int i = 0; i < 8; ++i) Bt[bcol + i][brow] = (bf16_t)wsrc[i];
    }
    if (A_BF16) wait_async0();
    __syncthreads();

    v16bf a[2], bfr[2];
    #pragma unroll
    for (int s = 0; s < 2; ++s) {
      const bf16_t* ap = &As[wm * 32 + s * 16 + lq][0];
      a[s] = cat8(*(const v8bf*)(ap + 8 * half), *(const v8bf*)(ap + 16 + 8 * half));
      const bf16_t* bp = &Bt[wn * 32 + s * 16 + lq][0];
      bfr[s] = cat8(*(const v8bf*)(bp + 8 * half), *(const v8bf*)(bp + 16 + 8 * half));
    }
    #pragma unroll
    for (int i = 0; i < 2; ++i)
      #pragma unroll
      for (int j = 0; j < 2; ++j)
        c[i][j] = __builtin_amdgcn_wmma_f32_16x16x32_bf16(
            false, a[i], false, bfr[j], (short)0, c[i][j], false, false);
    __syncthreads();
  }

  #pragma unroll
  for (int i = 0; i < 2; ++i)
    #pragma unroll
    for (int j = 0; j < 2; ++j)
      #pragma unroll
      for (int r = 0; r < 8; ++r) {
        const int row = m0 + wm * 32 + i * 16 + r + 8 * half;
        const int col = n0 + wn * 32 + j * 16 + lq;
        if (OUT_BF16) ((bf16_t*)Yv)[(size_t)row * N + col] = (bf16_t)c[i][j][r];
        else          ((float*)Yv)[(size_t)row * N + col]  = c[i][j][r];
      }
}

// ---------------------------------------------------------------------------
// Fused flash attention per (b, h, 128-q-row tile). 8 waves x 16 q rows.
// Computes S^T = K·Q^T (kv over VGPRs, q over lanes) so the probability
// C-fragments map directly onto the A-fragment of P·V (no shuffles).
// Online softmax with post-softmax multiplicative seq_mask on numerator.
// K chunk staged via async DMA to LDS; V transposed through VGPRs.
// ---------------------------------------------------------------------------
__global__ __launch_bounds__(256)
void attn_kernel(const bf16_t* __restrict__ Qp, const bf16_t* __restrict__ Kp,
                 const bf16_t* __restrict__ Vp, const float* __restrict__ att_mask,
                 const float* __restrict__ seq_mask, bf16_t* __restrict__ Op)
{
  __shared__ __align__(16) bf16_t Ks[32][72];   // K chunk [kv][d], padded
  __shared__ __align__(16) bf16_t Vt[64][40];   // V chunk transposed [d][kv], padded

  const int tid = threadIdx.x, lane = tid & 31, wave = tid >> 5;
  const int lq = lane & 15, half = lane >> 4;
  const int nqt = S_LEN / 128;
  const int qt = blockIdx.x % nqt;
  const int h  = (blockIdx.x / nqt) % NHEAD;
  const int b  = blockIdx.x / (nqt * NHEAD);

  const int q_local = qt * 128 + wave * 16 + lq;            // this lane's q row
  const size_t qrow = (size_t)b * S_LEN + q_local;

  // Q fragments (B-frag of S^T): table-order reads of the q row
  const bf16_t* qptr = Qp + qrow * D_MODEL + h * DK;
  const v16bf bq0 = cat8(*(const v8bf*)(qptr + 8 * half),      *(const v8bf*)(qptr + 16 + 8 * half));
  const v16bf bq1 = cat8(*(const v8bf*)(qptr + 32 + 8 * half), *(const v8bf*)(qptr + 48 + 8 * half));

  float mrow = -3.0e38f, Z = 0.0f;
  v8f acc[4] = {};
  const float* amrow = att_mask + ((size_t)b * S_LEN + q_local) * S_LEN;
  const float* smrow = seq_mask + ((size_t)b * S_LEN + q_local) * S_LEN;

  const int stg_kv = tid >> 3;        // 0..31
  const int stg_c  = (tid & 7) * 8;   // 0..56

  for (int kv0 = 0; kv0 < S_LEN; kv0 += 32) {
    // ---- stage K (async DMA, row-major) and V (transposed via VGPRs) ----
    {
      const size_t srow = ((size_t)b * S_LEN + kv0 + stg_kv) * D_MODEL + h * DK + stg_c;
      async_copy_b128(&Ks[stg_kv][stg_c], Kp + srow);
      const v8bf vx = *(const v8bf*)(Vp + srow);
      #pragma unroll
      for (int i = 0; i < 8; ++i) Vt[stg_c + i][stg_kv] = vx[i];
      if (kv0 + 32 < S_LEN) {   // prefetch next chunk + next mask segments
        const size_t nrow = srow + (size_t)32 * D_MODEL;
        __builtin_prefetch(Kp + nrow, 0, 0);
        __builtin_prefetch(Vp + nrow, 0, 0);
        __builtin_prefetch(amrow + kv0 + 32 + 8 * half, 0, 0);
        __builtin_prefetch(smrow + kv0 + 32 + 8 * half, 0, 0);
      }
    }
    wait_async0();
    __syncthreads();

    // ---- S^T chunk: two 16-kv subtiles, K(=d) = 64 -> 2 WMMA each ----
    v8f s[2];
    #pragma unroll
    for (int sub = 0; sub < 2; ++sub) {
      const bf16_t* kr = &Ks[sub * 16 + lq][0];
      const v16bf a0 = cat8(*(const v8bf*)(kr + 8 * half),      *(const v8bf*)(kr + 16 + 8 * half));
      const v16bf a1 = cat8(*(const v8bf*)(kr + 32 + 8 * half), *(const v8bf*)(kr + 48 + 8 * half));
      v8f cc = {};
      cc = __builtin_amdgcn_wmma_f32_16x16x32_bf16(false, a0, false, bq0, (short)0, cc, false, false);
      cc = __builtin_amdgcn_wmma_f32_16x16x32_bf16(false, a1, false, bq1, (short)0, cc, false, false);
      s[sub] = cc;
    }

    // ---- scale + att_mask, online softmax stats ----
    float sv[16];
    float cmax = -3.0e38f;
    const float* am0 = amrow + kv0 + 8 * half;   // frag0: kv = kv0 + 8*half + r
    const float* am1 = am0 + 16;                 // frag1: +16
    #pragma unroll
    for (int r = 0; r < 8; ++r) {
      const float x0 = s[0][r] * 0.125f + am0[r];
      const float x1 = s[1][r] * 0.125f + am1[r];
      sv[r] = x0; sv[8 + r] = x1;
      cmax = fmaxf(cmax, fmaxf(x0, x1));
    }
    cmax = fmaxf(cmax, __shfl_xor(cmax, 16, 32));   // combine wave halves (same q)
    const float newm  = fmaxf(mrow, cmax);
    const float alpha = __expf(mrow - newm);
    mrow = newm;

    const float* sm0 = smrow + kv0 + 8 * half;
    const float* sm1 = sm0 + 16;
    float zloc = 0.0f;
    float w[16];
    #pragma unroll
    for (int r = 0; r < 8; ++r) {
      const float p0 = __expf(sv[r] - mrow);
      const float p1 = __expf(sv[8 + r] - mrow);
      zloc += p0 + p1;
      w[r]     = p0 * sm0[r];          // post-softmax multiplicative mask
      w[8 + r] = p1 * sm1[r];
    }
    zloc += __shfl_xor(zloc, 16, 32);
    Z = Z * alpha + zloc;

    // ---- rescale accumulators: acc VGPR r holds q = r + 8*half ----
    #pragma unroll
    for (int r = 0; r < 8; ++r) {
      const float ar = __shfl(alpha, (half << 3) + r, 32);
      acc[0][r] *= ar; acc[1][r] *= ar; acc[2][r] *= ar; acc[3][r] *= ar;
    }

    // ---- probability A-fragment: direct C->A element mapping ----
    v16bf wf;
    #pragma unroll
    for (int e = 0; e < 16; ++e) wf[e] = (bf16_t)w[e];

    // ---- P·V: 4 x 16-wide d chunks, K(=kv) = 32 ----
    #pragma unroll
    for (int j = 0; j < 4; ++j) {
      const bf16_t* vr = &Vt[j * 16 + lq][0];
      const v16bf bv = cat8(*(const v8bf*)(vr + 8 * half), *(const v8bf*)(vr + 16 + 8 * half));
      acc[j] = __builtin_amdgcn_wmma_f32_16x16x32_bf16(false, wf, false, bv, (short)0, acc[j], false, false);
    }
    __syncthreads();
  }

  // ---- finalize: divide by softmax denominator, store bf16 head output ----
  const float rz = 1.0f / Z;
  #pragma unroll
  for (int r = 0; r < 8; ++r) {
    const float zr = __shfl(rz, (half << 3) + r, 32);
    const int srow = qt * 128 + wave * 16 + r + 8 * half;
    bf16_t* op = Op + ((size_t)b * S_LEN + srow) * D_MODEL + h * DK + lq;
    #pragma unroll
    for (int j = 0; j < 4; ++j) op[j * 16] = (bf16_t)(acc[j][r] * zr);
  }
}

// ---------------------------------------------------------------------------
extern "C" void kernel_launch(void* const* d_in, const int* in_sizes, int n_in,
                              void* d_out, int out_size, void* d_ws, size_t ws_size,
                              hipStream_t stream) {
  (void)in_sizes; (void)n_in; (void)out_size; (void)ws_size;
  // setup_inputs order: q, v, k, att_mask, seq_mask, Wq, Wk, Wv, Wo
  const float* q        = (const float*)d_in[0];
  const float* v        = (const float*)d_in[1];
  const float* k        = (const float*)d_in[2];
  const float* att_mask = (const float*)d_in[3];
  const float* seq_mask = (const float*)d_in[4];
  const float* Wq       = (const float*)d_in[5];
  const float* Wk       = (const float*)d_in[6];
  const float* Wv       = (const float*)d_in[7];
  const float* Wo       = (const float*)d_in[8];
  float* out = (float*)d_out;

  char* ws = (char*)d_ws;
  const size_t BUF = (size_t)MROWS * D_MODEL * sizeof(bf16_t);  // 8 MiB
  bf16_t* Qp = (bf16_t*)(ws);
  bf16_t* Kp = (bf16_t*)(ws + BUF);
  bf16_t* Vp = (bf16_t*)(ws + 2 * BUF);
  bf16_t* Op = (bf16_t*)(ws + 3 * BUF);

  dim3 gg(MROWS / 128, D_MODEL / 64);  // 32 x 16 blocks
  gemm_kernel<false, true><<<gg, 256, 0, stream>>>(q, Wq, Qp, MROWS, D_MODEL, D_MODEL);
  gemm_kernel<false, true><<<gg, 256, 0, stream>>>(k, Wk, Kp, MROWS, D_MODEL, D_MODEL);
  gemm_kernel<false, true><<<gg, 256, 0, stream>>>(v, Wv, Vp, MROWS, D_MODEL, D_MODEL);

  const int nblk = BATCH * NHEAD * (S_LEN / 128);  // 512
  attn_kernel<<<nblk, 256, 0, stream>>>(Qp, Kp, Vp, att_mask, seq_mask, Op);

  gemm_kernel<true, false><<<gg, 256, 0, stream>>>(Op, Wo, out, MROWS, D_MODEL, D_MODEL);
}
